// QuadraLayer_3985729651638
// MI455X (gfx1250) — matrix operand order repaired and checked by reading, not compiled, not run on hardware
//
#include <hip/hip_runtime.h>

// ---------------------------------------------------------------------------
// QuadraLayer for MI455X (gfx1250, wave32, WMMA).
// GEMMs (~86 GFLOP total) run on v_wmma_f32_16x16x32_f16 (f16 in, f32 acc),
// 2x2 sub-tiles per wave (4 WMMA / K-step), double-buffered LDS tiles
// (1 barrier / K-step). The two attentions have K-dims of 10-11 (< WMMA K=32)
// -> VALU + shuffle kernels.
// ---------------------------------------------------------------------------

typedef __attribute__((ext_vector_type(16))) _Float16 v16h;
typedef __attribute__((ext_vector_type(8)))  _Float16 f16x8;
typedef __attribute__((ext_vector_type(4)))  _Float16 f16x4;
typedef __attribute__((ext_vector_type(8)))  float    v8f;

#define T_SEQ   1024
#define L_DIM   32
#define D_MOD   128
#define M_ROWS  131072          // B*L*T = 4*32*1024
#define DFF     512
#define N_F     10
#define N_H     2
#define D_H     64
#define WINW    5
#define ATT_SCALE 0.125f        // Dh^-0.5, Dh=64
#define CLAMP_V 10000.0f
#define LN_EPS  1e-5f

__device__ __forceinline__ v16h cat8(f16x8 lo, f16x8 hi) {
  return __builtin_shufflevector(lo, hi, 0,1,2,3,4,5,6,7,8,9,10,11,12,13,14,15);
}
__device__ __forceinline__ float gelu_f(float v) {
  return 0.5f * v * (1.0f + erff(v * 0.7071067811865475f));
}
__device__ __forceinline__ v8f wmma_f16(v16h a, v16h b, v8f c) {
  return __builtin_amdgcn_wmma_f32_16x16x32_f16(false, a, false, b, (short)0, c, false, false);
}

// ---------------------------------------------------------------------------
// x [B,T,L,D] fp32  ->  x_dim [B*L*T, D] f16 (permuted)
// ---------------------------------------------------------------------------
__global__ __launch_bounds__(256)
void convert_x_kernel(const float* __restrict__ x, _Float16* __restrict__ xd) {
  size_t g = (size_t)blockIdx.x * 256 + threadIdx.x;
  size_t o = g * 4;                                   // flat [row, d]
  int    d   = (int)(o & (D_MOD - 1));
  size_t row = o >> 7;                                // bl*T + t
  int    t   = (int)(row & (T_SEQ - 1));
  size_t bl  = row >> 10;
  size_t b   = bl >> 5;
  int    l   = (int)(bl & (L_DIM - 1));
  const float* src = x + (((b * T_SEQ + t) * L_DIM + l) << 7) + d;
  float4 v = *(const float4*)src;
  f16x4 h;
  h[0] = (_Float16)v.x; h[1] = (_Float16)v.y;
  h[2] = (_Float16)v.z; h[3] = (_Float16)v.w;
  *(f16x4*)(xd + o) = h;
}

// router [L*F, D] fp32 -> f16
__global__ __launch_bounds__(256)
void convert_router_kernel(const float* __restrict__ r, _Float16* __restrict__ out) {
  int g = blockIdx.x * 256 + threadIdx.x;
  int o = g * 4;                                      // < 320*128
  float4 v = *(const float4*)(r + o);
  f16x4 h;
  h[0] = (_Float16)v.x; h[1] = (_Float16)v.y;
  h[2] = (_Float16)v.z; h[3] = (_Float16)v.w;
  *(f16x4*)(out + o) = h;
}

// ---------------------------------------------------------------------------
// C[M,N] = act(A[M,K](f16) @ W[K,N](f32->f16) + bias), f16 output.
// Block tile 128x64, 8 waves as 4(M)x2(N); each wave owns a 32x32 tile:
// 2 A-frags x 2 B-frags -> 4x v_wmma_f32_16x16x32_f16 per K-step.
// Double-buffered LDS: one barrier per K-step, global loads overlap WMMA.
// ---------------------------------------------------------------------------
__global__ __launch_bounds__(256)
void gemm_bias_act_f16(const _Float16* __restrict__ A, const float* __restrict__ W,
                       const float* __restrict__ bias, _Float16* __restrict__ C,
                       int M, int K, int N, int act)
{
  __shared__ __align__(16) _Float16 As[2][128][32];   // 2 x 8 KB
  __shared__ __align__(16) _Float16 Bs[2][32][64];    // 2 x 4 KB
  const int tid  = threadIdx.x;
  const int wave = tid >> 5, lane = tid & 31;
  const int lm = lane & 15, lh = lane >> 4;
  const int m0 = blockIdx.x * 128;
  const int n0 = blockIdx.y * 64;
  const int wm = wave & 3;        // 4 M-groups of 32 rows
  const int wn = wave >> 2;       // 2 N-groups of 32 cols
  const int Mlim = M - 1;

  v8f a00 = {}, a01 = {}, a10 = {}, a11 = {};

  auto stage = [&](int p, int k0) {
    { // A tile 128x32 f16: 16 halves/thread (2 x b128)
      int idx = tid << 4; int r = idx >> 5; int c = idx & 31;
      int mrow = m0 + r; if (mrow > Mlim) mrow = Mlim;      // clamp (M=320 case)
      const _Float16* src = A + (size_t)mrow * K + k0 + c;
      *(f16x8*)&As[p][r][c]     = *(const f16x8*)src;
      *(f16x8*)&As[p][r][c + 8] = *(const f16x8*)(src + 8);
    }
    { // B tile 32x64 with f32->f16 conversion: 8 floats/thread
      int idx = tid << 3; int r = idx >> 6; int c = idx & 63;
      const float* src = W + (size_t)(k0 + r) * N + n0 + c;
      f16x8 h;
      #pragma unroll
      for (int i = 0; i < 8; ++i) h[i] = (_Float16)src[i];
      *(f16x8*)&Bs[p][r][c] = h;
    }
  };

  auto compute = [&](int p) {
    const _Float16* ap0 = &As[p][(2 * wm + 0) * 16 + lm][lh * 8];
    const _Float16* ap1 = &As[p][(2 * wm + 1) * 16 + lm][lh * 8];
    v16h af0 = cat8(*(const f16x8*)ap0, *(const f16x8*)(ap0 + 16));
    v16h af1 = cat8(*(const f16x8*)ap1, *(const f16x8*)(ap1 + 16));
    v16h bf0, bf1;
    #pragma unroll
    for (int s = 0; s < 16; ++s) {
      int kk = (s < 8) ? (lh * 8 + s) : (16 + lh * 8 + (s - 8));
      bf0[s] = Bs[p][kk][(2 * wn + 0) * 16 + lm];
      bf1[s] = Bs[p][kk][(2 * wn + 1) * 16 + lm];
    }
    a00 = wmma_f16(af0, bf0, a00);
    a01 = wmma_f16(af0, bf1, a01);
    a10 = wmma_f16(af1, bf0, a10);
    a11 = wmma_f16(af1, bf1, a11);
  };

  stage(0, 0);
  __syncthreads();
  int p = 0;
  for (int k0 = 0; k0 < K; k0 += 32) {
    if (k0 + 32 < K) stage(p ^ 1, k0 + 32);
    compute(p);
    __syncthreads();
    p ^= 1;
  }

  // epilogue per 16x16 C layout: VGPR i -> M = (lane/16)*8 + i, N = lane%16
  #pragma unroll
  for (int mi = 0; mi < 2; ++mi) {
    #pragma unroll
    for (int nj = 0; nj < 2; ++nj) {
      v8f acc = mi ? (nj ? a11 : a10) : (nj ? a01 : a00);
      int n = n0 + (2 * wn + nj) * 16 + lm;
      float bv = bias[n];
      #pragma unroll
      for (int i = 0; i < 8; ++i) {
        int m = m0 + (2 * wm + mi) * 16 + lh * 8 + i;
        float v = acc[i] + bv;
        if (act) v = gelu_f(v);
        if (m < M) C[(size_t)m * N + n] = (_Float16)v;
      }
    }
  }
}

// ---------------------------------------------------------------------------
// Row-complete GEMM + residual + LayerNorm. N fixed = 128, block tile 64x128
// (8 waves as 2(M)x4(N), 2x2 sub-tiles each, double-buffered LDS).
//   mode 0: y = LN(res + A@W + b, ga, ba), write f32 + f16.
//   mode 1: z = LN(LN(res + A@W + b, ga, ba), gb, bb), permuted store (final).
// perm_res: residual read with [B,T,L,D] permutation (x_dim residual).
// ---------------------------------------------------------------------------
__global__ __launch_bounds__(256)
void gemm_bias_res_ln(const _Float16* __restrict__ A, const float* __restrict__ W,
                      const float* __restrict__ bias, const float* __restrict__ res,
                      int perm_res,
                      const float* __restrict__ ga, const float* __restrict__ ba,
                      const float* __restrict__ gb, const float* __restrict__ bb,
                      float* __restrict__ outf, _Float16* __restrict__ outh,
                      int mode, int K)
{
  __shared__ __align__(16) _Float16 As[2][64][32];    // 2 x 4 KB
  __shared__ __align__(16) _Float16 Bs[2][32][128];   // 2 x 8 KB
  __shared__ __align__(16) float    Cs[64][128];      // 32 KB
  __shared__ float mu_s[64];
  __shared__ float rs_s[64];
  const int tid  = threadIdx.x;
  const int wave = tid >> 5, lane = tid & 31;
  const int lm = lane & 15, lh = lane >> 4;
  const int m0 = blockIdx.x * 64;
  const int wm = wave & 1;        // 2 M-groups of 32 rows
  const int wn = wave >> 1;       // 4 N-groups of 32 cols

  v8f a00 = {}, a01 = {}, a10 = {}, a11 = {};

  auto stage = [&](int p, int k0) {
    { // A tile 64x32 f16: 8 halves/thread
      int idx = tid << 3; int r = idx >> 5; int c = idx & 31;
      *(f16x8*)&As[p][r][c] = *(const f16x8*)(A + (size_t)(m0 + r) * K + k0 + c);
    }
    { // B tile 32x128 with f32->f16 conversion: 16 floats/thread
      int idx = tid << 4; int r = idx >> 7; int c = idx & 127;
      const float* src = W + (size_t)(k0 + r) * D_MOD + c;
      f16x8 h0, h1;
      #pragma unroll
      for (int i = 0; i < 8; ++i) h0[i] = (_Float16)src[i];
      #pragma unroll
      for (int i = 0; i < 8; ++i) h1[i] = (_Float16)src[8 + i];
      *(f16x8*)&Bs[p][r][c]     = h0;
      *(f16x8*)&Bs[p][r][c + 8] = h1;
    }
  };

  auto compute = [&](int p) {
    const _Float16* ap0 = &As[p][(2 * wm + 0) * 16 + lm][lh * 8];
    const _Float16* ap1 = &As[p][(2 * wm + 1) * 16 + lm][lh * 8];
    v16h af0 = cat8(*(const f16x8*)ap0, *(const f16x8*)(ap0 + 16));
    v16h af1 = cat8(*(const f16x8*)ap1, *(const f16x8*)(ap1 + 16));
    v16h bf0, bf1;
    #pragma unroll
    for (int s = 0; s < 16; ++s) {
      int kk = (s < 8) ? (lh * 8 + s) : (16 + lh * 8 + (s - 8));
      bf0[s] = Bs[p][kk][(2 * wn + 0) * 16 + lm];
      bf1[s] = Bs[p][kk][(2 * wn + 1) * 16 + lm];
    }
    a00 = wmma_f16(af0, bf0, a00);
    a01 = wmma_f16(af0, bf1, a01);
    a10 = wmma_f16(af1, bf0, a10);
    a11 = wmma_f16(af1, bf1, a11);
  };

  stage(0, 0);
  __syncthreads();
  int p = 0;
  for (int k0 = 0; k0 < K; k0 += 32) {
    if (k0 + 32 < K) stage(p ^ 1, k0 + 32);
    compute(p);
    __syncthreads();
    p ^= 1;
  }

  // accumulators + bias -> LDS row buffer
  #pragma unroll
  for (int mi = 0; mi < 2; ++mi) {
    #pragma unroll
    for (int nj = 0; nj < 2; ++nj) {
      v8f acc = mi ? (nj ? a11 : a10) : (nj ? a01 : a00);
      int n = (2 * wn + nj) * 16 + lm;
      float bv = bias[n];
      #pragma unroll
      for (int i = 0; i < 8; ++i)
        Cs[(2 * wm + mi) * 16 + lh * 8 + i][n] = acc[i] + bv;
    }
  }
  __syncthreads();

  const int rr = tid >> 2;              // 0..63: row within block
  const int c0 = (tid & 3) << 5;        // 4 threads x 32 cols per row
  const size_t row = (size_t)m0 + rr;
  size_t roff;
  if (perm_res) {
    int t = (int)(row & (T_SEQ - 1));
    size_t bl = row >> 10;
    size_t b  = bl >> 5; int l = (int)(bl & 31);
    roff = ((b * T_SEQ + t) * L_DIM + l) << 7;
  } else {
    roff = row << 7;
  }
  #pragma unroll 8
  for (int i = 0; i < 32; ++i) Cs[rr][c0 + i] += res[roff + c0 + i];
  __syncthreads();

  if (tid < 64) {
    float s = 0.f, sq = 0.f;
    for (int c = 0; c < 128; ++c) { float v = Cs[tid][c]; s += v; sq += v * v; }
    float m = s * (1.f / 128.f);
    mu_s[tid] = m;
    rs_s[tid] = rsqrtf(sq * (1.f / 128.f) - m * m + LN_EPS);
  }
  __syncthreads();
  float m1 = mu_s[rr], r1 = rs_s[rr];

  if (mode == 0) {
    #pragma unroll 8
    for (int i = 0; i < 32; ++i) {
      int c = c0 + i;
      float y = (Cs[rr][c] - m1) * r1 * ga[c] + ba[c];
      outf[(row << 7) + c] = y;
      outh[(row << 7) + c] = (_Float16)y;
    }
  } else {
    #pragma unroll 8
    for (int i = 0; i < 32; ++i) {
      int c = c0 + i;
      Cs[rr][c] = (Cs[rr][c] - m1) * r1 * ga[c] + ba[c];
    }
    __syncthreads();
    if (tid < 64) {
      float s = 0.f, sq = 0.f;
      for (int c = 0; c < 128; ++c) { float v = Cs[tid][c]; s += v; sq += v * v; }
      float m = s * (1.f / 128.f);
      mu_s[tid] = m;
      rs_s[tid] = rsqrtf(sq * (1.f / 128.f) - m * m + LN_EPS);
    }
    __syncthreads();
    float m2 = mu_s[rr], r2 = rs_s[rr];
    int t = (int)(row & (T_SEQ - 1));
    size_t bl = row >> 10;
    size_t b  = bl >> 5; int l = (int)(bl & 31);
    size_t ooff = ((b * T_SEQ + t) * L_DIM + l) << 7;
    #pragma unroll 8
    for (int i = 0; i < 32; ++i) {
      int c = c0 + i;
      outf[ooff + c] = (Cs[rr][c] - m2) * r2 * gb[c] + bb[c];
    }
  }
}

// ---------------------------------------------------------------------------
// attention 1: router (F=10 queries) over K/V of length 1024, band +-5.
// One wave per (bl, h, f); <=11 dots of length 64 via shfl_xor reductions.
// ---------------------------------------------------------------------------
__global__ __launch_bounds__(32)
void attn1_kernel(const _Float16* __restrict__ Kp, const _Float16* __restrict__ Vp,
                  const _Float16* __restrict__ Rp, float* __restrict__ Buf)
{
  const int idx = blockIdx.x;
  const int f  = idx % N_F;
  const int h  = (idx / N_F) % N_H;
  const int bl = idx / (N_F * N_H);
  const int l  = bl & (L_DIM - 1);
  const int lane = threadIdx.x;
  const int d0 = lane << 1;
  const _Float16* rp = Rp + ((size_t)(l * N_F + f)) * D_MOD + h * D_H;
  const float q0 = (float)rp[d0], q1 = (float)rp[d0 + 1];
  const int jlo = (f - WINW > 0) ? (f - WINW) : 0;
  const int jhi = f + WINW;                   // <= 14 < T_SEQ, always in range
  float s[2 * WINW + 1];
  #pragma unroll
  for (int jj = 0; jj <= 2 * WINW; ++jj) {
    int j = jlo + jj;
    bool valid = (j <= jhi);
    const _Float16* kp = Kp + ((size_t)(bl * T_SEQ + j)) * D_MOD + h * D_H;
    float p = valid ? (q0 * (float)kp[d0] + q1 * (float)kp[d0 + 1]) : 0.f;
    #pragma unroll
    for (int m = 16; m > 0; m >>= 1) p += __shfl_xor(p, m, 32);
    float sc = fminf(fmaxf(p * ATT_SCALE, -CLAMP_V), CLAMP_V);
    s[jj] = valid ? sc : -1e30f;
  }
  float mx = -1e30f;
  #pragma unroll
  for (int jj = 0; jj <= 2 * WINW; ++jj) mx = fmaxf(mx, s[jj]);
  float Z = 0.f;
  #pragma unroll
  for (int jj = 0; jj <= 2 * WINW; ++jj) { float e = __expf(s[jj] - mx); s[jj] = e; Z += e; }
  float invZ = 1.f / Z;
  float o0 = 0.f, o1 = 0.f;
  #pragma unroll
  for (int jj = 0; jj <= 2 * WINW; ++jj) {
    const _Float16* vp = Vp + ((size_t)(bl * T_SEQ + jlo + jj)) * D_MOD + h * D_H;
    o0 += s[jj] * (float)vp[d0];
    o1 += s[jj] * (float)vp[d0 + 1];
  }
  float* out = Buf + ((size_t)((bl * N_H + h) * N_F + f)) * D_H + d0;
  out[0] = o0 * invZ;
  out[1] = o1 * invZ;
}

// ---------------------------------------------------------------------------
// attention 2: 1024 queries over 10 buffer slots (band only for t<10).
// Block = 8 waves = 8 consecutive t of one (bl,h); buffer tile in LDS.
// ---------------------------------------------------------------------------
__global__ __launch_bounds__(256)
void attn2_kernel(const _Float16* __restrict__ Qp, const float* __restrict__ Buf,
                  _Float16* __restrict__ Recv)
{
  __shared__ float Bls[N_F][D_H];   // 2.5 KB
  __shared__ float Qls[8][D_H];     // 2 KB
  const int tid  = threadIdx.x;
  const int wave = tid >> 5, lane = tid & 31;
  const int gid  = blockIdx.x;
  const int t8   = gid & 127;            // T/8 = 128 blocks per (bl,h)
  const int rest = gid >> 7;
  const int h  = rest & 1;
  const int bl = rest >> 1;
  const int t  = (t8 << 3) + wave;
  const int d0 = lane << 1;

  for (int i = tid; i < N_F * D_H; i += 256)
    Bls[i / D_H][i % D_H] = Buf[((size_t)(bl * N_H + h) * N_F) * D_H + i];
  const _Float16* q = Qp + ((size_t)(bl * T_SEQ + t)) * D_MOD + h * D_H;
  Qls[wave][d0]     = (float)q[d0];
  Qls[wave][d0 + 1] = (float)q[d0 + 1];
  __syncthreads();

  float sj = -1e30f;
  if (lane < N_F) {
    float a = 0.f;
    #pragma unroll 8
    for (int d = 0; d < D_H; ++d) a += Qls[wave][d] * Bls[lane][d];
    a = fminf(fmaxf(a * ATT_SCALE, -CLAMP_V), CLAMP_V);
    if (t < N_F && (lane < t - WINW || lane > t + WINW)) a = -1e30f;  // band rows
    sj = a;
  }
  float s[N_F];
  #pragma unroll
  for (int j = 0; j < N_F; ++j) s[j] = __shfl(sj, j, 32);
  float mx = -1e30f;
  #pragma unroll
  for (int j = 0; j < N_F; ++j) mx = fmaxf(mx, s[j]);
  float Z = 0.f;
  #pragma unroll
  for (int j = 0; j < N_F; ++j) { float e = __expf(s[j] - mx); s[j] = e; Z += e; }
  float invZ = 1.f / Z;
  float o0 = 0.f, o1 = 0.f;
  #pragma unroll
  for (int j = 0; j < N_F; ++j) { o0 += s[j] * Bls[j][d0]; o1 += s[j] * Bls[j][d0 + 1]; }
  _Float16* out = Recv + ((size_t)(bl * T_SEQ + t)) * D_MOD + h * D_H + d0;
  out[0] = (_Float16)(o0 * invZ);
  out[1] = (_Float16)(o1 * invZ);
}

// ---------------------------------------------------------------------------
extern "C" void kernel_launch(void* const* d_in, const int* in_sizes, int n_in,
                              void* d_out, int out_size, void* d_ws, size_t ws_size,
                              hipStream_t stream) {
  (void)in_sizes; (void)n_in; (void)out_size; (void)ws_size;
  const float* x        = (const float*)d_in[0];
  const float* router   = (const float*)d_in[1];
  const float* w_router = (const float*)d_in[2];
  const float* b_router = (const float*)d_in[3];
  const float* w_k  = (const float*)d_in[4];
  const float* b_k  = (const float*)d_in[5];
  const float* w_v  = (const float*)d_in[6];
  const float* b_v  = (const float*)d_in[7];
  const float* w_q  = (const float*)d_in[8];
  const float* b_q  = (const float*)d_in[9];
  const float* m1w1 = (const float*)d_in[10];
  const float* m1b1 = (const float*)d_in[11];
  const float* m1w2 = (const float*)d_in[12];
  const float* m1b2 = (const float*)d_in[13];
  const float* m2w1 = (const float*)d_in[14];
  const float* m2b1 = (const float*)d_in[15];
  const float* m2w2 = (const float*)d_in[16];
  const float* m2b2 = (const float*)d_in[17];
  const float* g1  = (const float*)d_in[18];
  const float* be1 = (const float*)d_in[19];
  const float* g2  = (const float*)d_in[20];
  const float* be2 = (const float*)d_in[21];
  const float* g3  = (const float*)d_in[22];
  const float* be3 = (const float*)d_in[23];

  char* ws = (char*)d_ws;
  size_t off = 0;
  auto wsalloc = [&](size_t bytes) {
    char* p = ws + off;
    off += (bytes + 255) & ~(size_t)255;
    return p;
  };
  _Float16* xd_f16  = (_Float16*)wsalloc((size_t)M_ROWS * D_MOD * 2);   // dead after mlp1-up
  float*    xproj   = (float*)   wsalloc((size_t)M_ROWS * D_MOD * 4);
  _Float16* xproj_h = (_Float16*)wsalloc((size_t)M_ROWS * D_MOD * 2);
  _Float16* h_f16   = (_Float16*)wsalloc((size_t)M_ROWS * DFF   * 2);   // shared by both MLPs
  _Float16* q_f16   = (_Float16*)wsalloc((size_t)M_ROWS * D_MOD * 2);
  _Float16* k_f16   = (_Float16*)wsalloc((size_t)M_ROWS * D_MOD * 2);
  _Float16* v_f16   = (_Float16*)wsalloc((size_t)M_ROWS * D_MOD * 2);
  _Float16* rA_f16  = (_Float16*)wsalloc((size_t)L_DIM * N_F * D_MOD * 2);
  _Float16* rproj_h = (_Float16*)wsalloc((size_t)L_DIM * N_F * D_MOD * 2);
  float*    buf     = (float*)   wsalloc((size_t)128 * N_H * N_F * D_H * 4);
  _Float16* recv_f16 = xd_f16;   // alias: x_dim f16 is dead after mlp1-up GEMM

  // x permute + f16 convert
  convert_x_kernel<<<16384, 256, 0, stream>>>(x, xd_f16);
  // mlp1 up: h = gelu(x_dim @ W1 + b1)   [M,128]@[128,512]
  gemm_bias_act_f16<<<dim3(M_ROWS / 128, DFF / 64), 256, 0, stream>>>(
      xd_f16, m1w1, m1b1, h_f16, M_ROWS, D_MOD, DFF, 1);
  // x_proj = LN(x_dim + h @ W2 + b2)     [M,512]@[512,128], permuted residual
  gemm_bias_res_ln<<<M_ROWS / 64, 256, 0, stream>>>(
      h_f16, m1w2, m1b2, x, 1, g1, be1, g1, be1, xproj, xproj_h, 0, DFF);
  // Q / K / V projections  [M,128]@[128,128]
  gemm_bias_act_f16<<<dim3(M_ROWS / 128, 2), 256, 0, stream>>>(
      xproj_h, w_q, b_q, q_f16, M_ROWS, D_MOD, D_MOD, 0);
  gemm_bias_act_f16<<<dim3(M_ROWS / 128, 2), 256, 0, stream>>>(
      xproj_h, w_k, b_k, k_f16, M_ROWS, D_MOD, D_MOD, 0);
  gemm_bias_act_f16<<<dim3(M_ROWS / 128, 2), 256, 0, stream>>>(
      xproj_h, w_v, b_v, v_f16, M_ROWS, D_MOD, D_MOD, 0);
  // router projection (batch-invariant: only L*F = 320 rows)
  convert_router_kernel<<<40, 256, 0, stream>>>(router, rA_f16);
  gemm_bias_act_f16<<<dim3(3, 2), 256, 0, stream>>>(
      rA_f16, w_router, b_router, rproj_h, 320, D_MOD, D_MOD, 0);
  // banded router attention -> buffer [BL,H,F,Dh]
  attn1_kernel<<<128 * N_H * N_F, 32, 0, stream>>>(k_f16, v_f16, rproj_h, buf);
  // queries over buffer -> recv [M,128] f16
  attn2_kernel<<<128 * N_H * (T_SEQ / 8), 256, 0, stream>>>(q_f16, buf, recv_f16);
  // mlp2 up
  gemm_bias_act_f16<<<dim3(M_ROWS / 128, DFF / 64), 256, 0, stream>>>(
      recv_f16, m2w1, m2b1, h_f16, M_ROWS, D_MOD, DFF, 1);
  // final: LN(LN(x_proj + h @ W2 + b2, g2, be2), g3, be3) -> d_out (permuted)
  gemm_bias_res_ln<<<M_ROWS / 64, 256, 0, stream>>>(
      h_f16, m2w2, m2b2, xproj, 0, g2, be2, g3, be3,
      (float*)d_out, xproj_h, 1, DFF);
}
